// ContextMessageBlock_80616536146580
// MI455X (gfx1250) — compile-verified, matrix operand order checked by reading, not run on hardware
//
#include <hip/hip_runtime.h>
#include <hip/hip_bf16.h>

typedef __attribute__((ext_vector_type(16))) _Float16 v16h;
typedef __attribute__((ext_vector_type(8)))  float    v8f;

#define HIDN 256
#define RBF  32
#define KIN1 800   // 3*HIDN + RBF
#define KINU 512   // 2*HIDN
#define EROWS 64   // edges per block (4 x 16-row WMMA sub-tiles)
#define ASTRIDE 816  // padded halfs per A row
#define NNODE 20000
#define NEDGE 320000

__device__ __forceinline__ float silu_f(float x) {
    return x / (1.0f + __expf(-x));
}

// Load one 16-half WMMA fragment slice: halves [p..p+7] and [p+16..p+23].
// Caller passes p = base + k0 + group*8 (16-byte aligned).
__device__ __forceinline__ v16h frag16(const _Float16* p) {
    union { v16h v; uint4 u[2]; } r;
    r.u[0] = *(const uint4*)(p);
    r.u[1] = *(const uint4*)(p + 16);
    return r.v;
}

#define WMMA_F16(a, b, c) \
    __builtin_amdgcn_wmma_f32_16x16x32_f16(false, (a), false, (b), (short)0, (c), false, false)

// ---------------- prep kernels ----------------

__global__ void k_zero_f32(float* p, int n) {
    int i = blockIdx.x * blockDim.x + threadIdx.x;
    if (i < n) p[i] = 0.0f;
}

__global__ void k_cvt_f16(const float* __restrict__ src, _Float16* __restrict__ dst, int n) {
    int i = blockIdx.x * blockDim.x + threadIdx.x;
    if (i < n) dst[i] = (_Float16)src[i];
}

// src is [K][Ncol] f32 row-major -> dst [Ncol][K] f16 (so each output column is
// a contiguous K-line; B fragments become two contiguous 16B loads per lane).
__global__ void k_transpose_f16(const float* __restrict__ src, _Float16* __restrict__ dst,
                                int K, int Ncol) {
    int i = blockIdx.x * blockDim.x + threadIdx.x;
    if (i >= K * Ncol) return;
    int k = i / Ncol, n = i % Ncol;
    dst[(size_t)n * K + k] = (_Float16)src[(size_t)k * Ncol + n];
}

__global__ void k_count(const int* __restrict__ ei, float* __restrict__ cnt, int E) {
    int e = blockIdx.x * blockDim.x + threadIdx.x;
    if (e < E) unsafeAtomicAdd(&cnt[ei[E + e]], 1.0f);
}

// ---------------- edge message kernel ----------------
// 512 threads = 16 waves; wave w owns output columns [16w, 16w+16).
// Block handles EROWS=64 edges (four 16-row WMMA sub-tiles sharing each
// B fragment -> 4 WMMAs per weight-fragment load, halved L2 weight traffic).
__launch_bounds__(512, 1)
__global__ void k_edge(const _Float16* __restrict__ hf16,
                       const float*    __restrict__ pos,
                       const int*      __restrict__ ei,     // [2E]
                       const int*      __restrict__ etype,  // [E]
                       const _Float16* __restrict__ embf,   // [2][256]
                       const _Float16* __restrict__ w1t,    // [256][800]
                       const float*    __restrict__ b1,
                       const _Float16* __restrict__ w2t,    // [256][256]
                       const float*    __restrict__ b2,
                       float*          __restrict__ seg,    // [N][256]
                       int E)
{
    __shared__ _Float16 A[EROWS][ASTRIDE];   // 64 x 800 (+pad) f16 input tile (~104KB)
    __shared__ _Float16 Mt[EROWS][HIDN];     // 64 x 256 f16 intermediate (32KB)
    __shared__ int   sSrc[EROWS], sDst[EROWS];
    __shared__ float sDist[EROWS];

    const int tid = threadIdx.x;
    const int e0  = blockIdx.x * EROWS;

    if (tid < EROWS)            sSrc[tid]         = ei[e0 + tid];
    else if (tid < 2 * EROWS)   sDst[tid - EROWS] = ei[E + e0 + tid - EROWS];
    __syncthreads();

    // stage h[src] | h[dst] | emb[etype]  (uint4 = 8 halfs per chunk)
    for (int i = tid; i < EROWS * 32; i += 512) {
        int r = i >> 5, c = i & 31;
        ((uint4*)&A[r][0])[c] = ((const uint4*)(hf16 + (size_t)sSrc[r] * HIDN))[c];
    }
    for (int i = tid; i < EROWS * 32; i += 512) {
        int r = i >> 5, c = i & 31;
        ((uint4*)&A[r][HIDN])[c] = ((const uint4*)(hf16 + (size_t)sDst[r] * HIDN))[c];
    }
    for (int i = tid; i < EROWS * 32; i += 512) {
        int r = i >> 5, c = i & 31;
        ((uint4*)&A[r][2 * HIDN])[c] = ((const uint4*)(embf + (size_t)etype[e0 + r] * HIDN))[c];
    }
    if (tid < EROWS) {
        int s = sSrc[tid], d = sDst[tid];
        float dx = pos[s * 3 + 0] - pos[d * 3 + 0];
        float dy = pos[s * 3 + 1] - pos[d * 3 + 1];
        float dz = pos[s * 3 + 2] - pos[d * 3 + 2];
        sDist[tid] = sqrtf(dx * dx + dy * dy + dz * dz);
    }
    __syncthreads();

    // radial RBF features: centers = linspace(0, 6, 32)
    const float STEP  = 6.0f / 31.0f;
    const float GAMMA = 1.0f / (STEP * STEP);
    for (int i = tid; i < EROWS * RBF; i += 512) {
        int r = i >> 5, c = i & 31;
        float diff = sDist[r] - (float)c * STEP;
        A[r][3 * HIDN + c] = (_Float16)__expf(-GAMMA * diff * diff);
    }
    __syncthreads();

    const int wave = tid >> 5;
    const int lane = tid & 31;
    const int ln   = lane & 15;
    const int grp  = lane >> 4;
    const int col  = wave * 16 + ln;

    // ---- GEMM1: [64 x 800] x [800 x 16] + b1, f32 accumulate ----
    v8f acc0, acc1, acc2, acc3;
    {
        float bv = b1[col];
        for (int r = 0; r < 8; r++) { acc0[r] = bv; acc1[r] = bv; acc2[r] = bv; acc3[r] = bv; }
    }
    {
        const _Float16* bp  = w1t + (size_t)col * KIN1;
        const _Float16* a0p = &A[ln][0];
        const _Float16* a1p = &A[16 + ln][0];
        const _Float16* a2p = &A[32 + ln][0];
        const _Float16* a3p = &A[48 + ln][0];
        for (int k0 = 0; k0 < KIN1; k0 += 32) {
            int ko = k0 + grp * 8;
            v16h bf = frag16(bp + ko);
            acc0 = WMMA_F16(frag16(a0p + ko), bf, acc0);
            acc1 = WMMA_F16(frag16(a1p + ko), bf, acc1);
            acc2 = WMMA_F16(frag16(a2p + ko), bf, acc2);
            acc3 = WMMA_F16(frag16(a3p + ko), bf, acc3);
        }
    }
    // SiLU, repack D (VGPR r = rows r / r+8, N = lane&15) to f16 LDS
    for (int r = 0; r < 8; r++) {
        int row = r + grp * 8;
        Mt[row][col]      = (_Float16)silu_f(acc0[r]);
        Mt[16 + row][col] = (_Float16)silu_f(acc1[r]);
        Mt[32 + row][col] = (_Float16)silu_f(acc2[r]);
        Mt[48 + row][col] = (_Float16)silu_f(acc3[r]);
    }
    __syncthreads();

    // ---- GEMM2: [64 x 256] x [256 x 16] + b2 ----
    v8f c0, c1, c2, c3;
    {
        float bv = b2[col];
        for (int r = 0; r < 8; r++) { c0[r] = bv; c1[r] = bv; c2[r] = bv; c3[r] = bv; }
    }
    {
        const _Float16* bp  = w2t + (size_t)col * HIDN;
        const _Float16* a0p = &Mt[ln][0];
        const _Float16* a1p = &Mt[16 + ln][0];
        const _Float16* a2p = &Mt[32 + ln][0];
        const _Float16* a3p = &Mt[48 + ln][0];
        for (int k0 = 0; k0 < HIDN; k0 += 32) {
            int ko = k0 + grp * 8;
            v16h bf = frag16(bp + ko);
            c0 = WMMA_F16(frag16(a0p + ko), bf, c0);
            c1 = WMMA_F16(frag16(a1p + ko), bf, c1);
            c2 = WMMA_F16(frag16(a2p + ko), bf, c2);
            c3 = WMMA_F16(frag16(a3p + ko), bf, c3);
        }
    }
    // SiLU + scatter-add m_ij into seg[dst]
    for (int r = 0; r < 8; r++) {
        int row = r + grp * 8;
        unsafeAtomicAdd(&seg[(size_t)sDst[row]      * HIDN + col], silu_f(c0[r]));
        unsafeAtomicAdd(&seg[(size_t)sDst[16 + row] * HIDN + col], silu_f(c1[r]));
        unsafeAtomicAdd(&seg[(size_t)sDst[32 + row] * HIDN + col], silu_f(c2[r]));
        unsafeAtomicAdd(&seg[(size_t)sDst[48 + row] * HIDN + col], silu_f(c3[r]));
    }
}

// ---------------- node update kernel ----------------
// 512 threads = 16 waves; 16 nodes per block.
__launch_bounds__(512, 1)
__global__ void k_node(const _Float16* __restrict__ hf16,
                       const float*    __restrict__ seg,    // [N][256] summed messages
                       const float*    __restrict__ cnt,    // [N]
                       const _Float16* __restrict__ w1t,    // [256][512]
                       const float*    __restrict__ b1,
                       const _Float16* __restrict__ w2t,    // [256][256]
                       const float*    __restrict__ b2,
                       float*          __restrict__ upd)    // [N][256]
{
    __shared__ _Float16 A[16][KINU];    // 16 x 512 f16: [h | m_i]
    __shared__ _Float16 Ut[16][HIDN];   // 16 x 256 f16 intermediate

    const int tid = threadIdx.x;
    const int n0  = blockIdx.x * 16;

    // stage h rows (512 uint4 chunks)
    {
        int r = tid >> 5, c = tid & 31;
        ((uint4*)&A[r][0])[c] = ((const uint4*)(hf16 + (size_t)(n0 + r) * HIDN))[c];
    }
    // stage m_i = seg / max(cnt, 1): 8 contiguous f32 per thread
    {
        int r = tid >> 5, c0 = (tid & 31) * 8;
        int n = n0 + r;
        float inv = 1.0f / fmaxf(cnt[n], 1.0f);
        for (int j = 0; j < 8; j++)
            A[r][HIDN + c0 + j] = (_Float16)(seg[(size_t)n * HIDN + c0 + j] * inv);
    }
    __syncthreads();

    const int wave = tid >> 5;
    const int lane = tid & 31;
    const int ln   = lane & 15;
    const int grp  = lane >> 4;
    const int col  = wave * 16 + ln;

    // ---- GEMM1: [16 x 512] x [512 x 16] + b1u ----
    v8f acc;
    {
        float bv = b1[col];
        for (int r = 0; r < 8; r++) acc[r] = bv;
    }
    {
        const _Float16* bp = w1t + (size_t)col * KINU;
        const _Float16* ap = &A[ln][0];
        for (int k0 = 0; k0 < KINU; k0 += 32) {
            int ko = k0 + grp * 8;
            acc = WMMA_F16(frag16(ap + ko), frag16(bp + ko), acc);
        }
    }
    for (int r = 0; r < 8; r++)
        Ut[r + grp * 8][col] = (_Float16)silu_f(acc[r]);
    __syncthreads();

    // ---- GEMM2: [16 x 256] x [256 x 16] + b2u ----
    v8f c2;
    {
        float bv = b2[col];
        for (int r = 0; r < 8; r++) c2[r] = bv;
    }
    {
        const _Float16* bp = w2t + (size_t)col * HIDN;
        const _Float16* ap = &Ut[ln][0];
        for (int k0 = 0; k0 < HIDN; k0 += 32) {
            int ko = k0 + grp * 8;
            c2 = WMMA_F16(frag16(ap + ko), frag16(bp + ko), c2);
        }
    }
    for (int r = 0; r < 8; r++) {
        int row = r + grp * 8;
        upd[(size_t)(n0 + row) * HIDN + col] = c2[r];
    }
}

// ---------------- layernorm / select kernel ----------------
// 256 threads = 8 waves; one wave per node row.
__launch_bounds__(256, 2)
__global__ void k_ln(const float* __restrict__ h,
                     const float* __restrict__ upd,
                     const int*   __restrict__ ntype,
                     const float* __restrict__ g,
                     const float* __restrict__ b,
                     float*       __restrict__ out)
{
    const int n    = blockIdx.x * 8 + (threadIdx.x >> 5);
    const int lane = threadIdx.x & 31;

    float x[8], hh[8];
    float sum = 0.0f;
    for (int j = 0; j < 8; j++) {
        int c = lane + j * 32;
        hh[j] = h[(size_t)n * HIDN + c];
        x[j]  = hh[j] + upd[(size_t)n * HIDN + c];
        sum  += x[j];
    }
    for (int off = 16; off > 0; off >>= 1) sum += __shfl_xor(sum, off, 32);
    float mu = sum * (1.0f / HIDN);

    float vs = 0.0f;
    for (int j = 0; j < 8; j++) { float d = x[j] - mu; vs += d * d; }
    for (int off = 16; off > 0; off >>= 1) vs += __shfl_xor(vs, off, 32);
    float inv = rsqrtf(vs * (1.0f / HIDN) + 1e-5f);

    bool lig = (ntype[n] == 1);
    for (int j = 0; j < 8; j++) {
        int c = lane + j * 32;
        out[(size_t)n * HIDN + c] = lig ? ((x[j] - mu) * inv * g[c] + b[c]) : hh[j];
    }
}

// ---------------- launch ----------------

extern "C" void kernel_launch(void* const* d_in, const int* in_sizes, int n_in,
                              void* d_out, int out_size, void* d_ws, size_t ws_size,
                              hipStream_t stream) {
    const float* h    = (const float*)d_in[0];
    const float* pos  = (const float*)d_in[1];
    const int*   ei   = (const int*)d_in[2];
    const int*   et   = (const int*)d_in[3];
    const int*   nt   = (const int*)d_in[4];
    const float* emb  = (const float*)d_in[5];
    const float* W1m  = (const float*)d_in[6];
    const float* b1m  = (const float*)d_in[7];
    const float* W2m  = (const float*)d_in[8];
    const float* b2m  = (const float*)d_in[9];
    const float* W1u  = (const float*)d_in[10];
    const float* b1u  = (const float*)d_in[11];
    const float* W2u  = (const float*)d_in[12];
    const float* b2u  = (const float*)d_in[13];
    const float* lng  = (const float*)d_in[14];
    const float* lnb  = (const float*)d_in[15];
    float* out = (float*)d_out;

    // workspace carve-out (256B aligned)
    char* ws = (char*)d_ws;
    size_t off = 0;
    auto take = [&](size_t bytes) -> char* {
        char* p = ws + off;
        off = (off + bytes + 255) & ~(size_t)255;
        return p;
    };
    _Float16* hf16 = (_Float16*)take((size_t)NNODE * HIDN * 2);
    _Float16* w1mt = (_Float16*)take((size_t)HIDN * KIN1 * 2);
    _Float16* w2mt = (_Float16*)take((size_t)HIDN * HIDN * 2);
    _Float16* w1ut = (_Float16*)take((size_t)HIDN * KINU * 2);
    _Float16* w2ut = (_Float16*)take((size_t)HIDN * HIDN * 2);
    _Float16* embf = (_Float16*)take((size_t)2 * HIDN * 2);
    float*    seg  = (float*)take((size_t)NNODE * HIDN * 4);
    float*    cnt  = (float*)take((size_t)NNODE * 4);
    float*    upd  = seg;  // aliased: each node block reads its seg rows into
                           // LDS before overwriting the same rows with upd.

    auto blk = [](int n, int t) { return (n + t - 1) / t; };

    // prep
    k_zero_f32<<<blk(NNODE * HIDN, 256), 256, 0, stream>>>(seg, NNODE * HIDN);
    k_zero_f32<<<blk(NNODE, 256), 256, 0, stream>>>(cnt, NNODE);
    k_cvt_f16<<<blk(NNODE * HIDN, 256), 256, 0, stream>>>(h, hf16, NNODE * HIDN);
    k_cvt_f16<<<blk(2 * HIDN, 256), 256, 0, stream>>>(emb, embf, 2 * HIDN);
    k_transpose_f16<<<blk(KIN1 * HIDN, 256), 256, 0, stream>>>(W1m, w1mt, KIN1, HIDN);
    k_transpose_f16<<<blk(HIDN * HIDN, 256), 256, 0, stream>>>(W2m, w2mt, HIDN, HIDN);
    k_transpose_f16<<<blk(KINU * HIDN, 256), 256, 0, stream>>>(W1u, w1ut, KINU, HIDN);
    k_transpose_f16<<<blk(HIDN * HIDN, 256), 256, 0, stream>>>(W2u, w2ut, HIDN, HIDN);
    k_count<<<blk(NEDGE, 256), 256, 0, stream>>>(ei, cnt, NEDGE);

    // edge messages (dominant GEMMs) + scatter
    k_edge<<<NEDGE / EROWS, 512, 0, stream>>>(hf16, pos, ei, et, embf,
                                              w1mt, b1m, w2mt, b2m, seg, NEDGE);
    // node update GEMMs
    k_node<<<NNODE / 16, 512, 0, stream>>>(hf16, seg, cnt, w1ut, b1u, w2ut, b2u, upd);
    // residual + layernorm + ligand select
    k_ln<<<NNODE / 8, 256, 0, stream>>>(h, upd, nt, lng, lnb, out);
}
